// MSAModule_89910845374475
// MI455X (gfx1250) — compile-verified, matrix operand order checked by reading, not run on hardware
//
#include <hip/hip_runtime.h>
#include <hip/hip_bf16.h>

// ====================== CDNA5 WMMA types & helpers ======================
typedef __attribute__((ext_vector_type(16))) __bf16 v16bf;
typedef __attribute__((ext_vector_type(8)))  float  v8f;
typedef __attribute__((__vector_size__(16))) int    v4i;
typedef __attribute__((address_space(1))) v4i* gas_v4i;   // global addrspace ptr
typedef __attribute__((address_space(3))) v4i* las_v4i;   // LDS addrspace ptr

#if defined(__has_builtin)
#if __has_builtin(__builtin_amdgcn_global_load_async_to_lds_b128)
#define HAVE_ASYNC_LDS 1
#else
#define HAVE_ASYNC_LDS 0
#endif
#else
#define HAVE_ASYNC_LDS 0
#endif

__device__ __forceinline__ v8f wmma_bf16(v16bf a, v16bf b, v8f c) {
  // v_wmma_f32_16x16x32_bf16: D = A(16x32) * B(32x16) + C(16x16 f32)
  return __builtin_amdgcn_wmma_f32_16x16x32_bf16(false, a, false, b, (short)0, c,
                                                 false, false);
}

__device__ __forceinline__ unsigned short f2bf(float f) {
  union { float f; unsigned u; } v; v.f = f;
  unsigned r = v.u + 0x7FFFu + ((v.u >> 16) & 1u);   // round-to-nearest-even
  return (unsigned short)(r >> 16);
}

// Index into a 16-bit WMMA operand fragment buffer laid out as
//   [kTile][mTile][lane(32)][halfword(16)]    (tile = 512 halfwords)
// Per ISA 7.12.2 (16-bit A 16x32): lane = m%16 + 16*bit3(k%32),
//   hw = (k&7) + 8*bit4(k%32). B uses the mirrored layout with m = N column.
__device__ __forceinline__ size_t fragIdx(int mTilesTotal, int k, int m) {
  int kT = k >> 5, ki = k & 31;
  int mT = m >> 4, mi = m & 15;
  int lane = mi + (((ki >> 3) & 1) << 4);
  int hw   = (ki & 7) + ((ki >> 4) << 3);
  return (((size_t)kT * mTilesTotal + mT) << 9) + (size_t)(lane << 4) + (size_t)hw;
}

// ====================== problem dims ======================
#define S_DIM   512
#define N_DIM   256
#define CM      64
#define CZ      128
#define COPM    32
#define NHEAD   8
#define CHD     32
#define NROWS   (S_DIM * N_DIM)          // 131072  (s,n) rows of m

// ====================== workspace layout (bytes) ======================
// all offsets 256B aligned; total = 238,649,344 bytes (~228 MiB)
static const size_t OFF_AFRAG = 0;                        // 512x8192 bf16  (8 MiB)
static const size_t OFF_BFRAG = OFF_AFRAG + 8388608;      // 512x8192 bf16  (8 MiB)
static const size_t OFF_MN2F  = OFF_BFRAG + 8388608;      // 131072x64 bf16 (16 MiB)
static const size_t OFF_OWF   = OFF_MN2F  + 16777216;     // 1024x128 bf16  (256 KiB)
static const size_t OFF_MGW   = OFF_OWF   + 262144;       // 64x512 bf16    (64 KiB)
static const size_t OFF_POW   = OFF_MGW   + 65536;        // 256x64 bf16    (32 KiB)
static const size_t OFF_RNUM  = OFF_POW   + 32768;        // 256x256 f32    (256 KiB)
static const size_t OFF_BIAS  = OFF_RNUM  + 262144;       // 8x256x256 f32  (2 MiB)
static const size_t OFF_WFRG  = OFF_BIAS  + 2097152;      // 8x 256x256 bf16(1 MiB)
static const size_t OFF_VFRG  = OFF_WFRG  + 1048576;      // 8x 256x16384   (64 MiB)
static const size_t OFF_GFRG  = OFF_VFRG  + 67108864;     // 131072x256     (64 MiB)
static const size_t OFF_OFRG  = OFF_GFRG  + 67108864;     // 131072x256     (64 MiB)
static const size_t WS_TOTAL  = OFF_OFRG  + 67108864;

// ====================== k_prep: weights -> bf16 B-fragments ======================
__global__ void __launch_bounds__(256)
k_prep(const float* __restrict__ opm_o_w, const float* __restrict__ pwa_m_w,
       const float* __restrict__ pwa_g_w, const float* __restrict__ pwa_o_w,
       unsigned short* __restrict__ owf, unsigned short* __restrict__ mgw,
       unsigned short* __restrict__ powf) {
  int id = blockIdx.x * 256 + threadIdx.x;
  if (id < 128 * 1024) {                       // opm_o_w[z][cd] : K=cd(1024), N=z(128)
    int z = id >> 10, k = id & 1023;
    owf[fragIdx(8, k, z)] = f2bf(opm_o_w[id]);
    return;
  }
  id -= 128 * 1024;
  if (id < 512 * 64) {                         // [pwa_m_w | pwa_g_w] : K=64, N=512
    int o = id >> 6, k = id & 63;
    float w = (o < 256) ? pwa_m_w[o * 64 + k] : pwa_g_w[(o - 256) * 64 + k];
    mgw[fragIdx(32, k, o)] = f2bf(w);
    return;
  }
  id -= 512 * 64;
  if (id < 64 * 256) {                         // pwa_o_w[out][in] : K=256, N=64
    int o = id >> 8, k = id & 255;
    powf[fragIdx(4, k, o)] = f2bf(pwa_o_w[id]);
  }
}

// ====================== k_num: 1 / clip(sum_s mask_si*mask_sj, 1) ======================
__global__ void __launch_bounds__(256)
k_num(const float* __restrict__ msa_mask, float* __restrict__ rnum) {
  int id = blockIdx.x * 256 + threadIdx.x;     // 65536 = i*256+j
  int i = id >> 8, j = id & 255;
  float s = 0.f;
  for (int t = 0; t < S_DIM; ++t)
    s += msa_mask[t * N_DIM + i] * msa_mask[t * N_DIM + j];
  rnum[id] = __builtin_amdgcn_rcpf(fmaxf(s, 1.0f));
}

// ====================== k_stage_m: layernorms + a/b projections ======================
__global__ void __launch_bounds__(256)
k_stage_m(const float* __restrict__ m, const float* __restrict__ msa_mask,
          const float* __restrict__ onw, const float* __restrict__ onb,
          const float* __restrict__ oaw, const float* __restrict__ obw,
          const float* __restrict__ pnw, const float* __restrict__ pnb,
          unsigned short* __restrict__ afrag, unsigned short* __restrict__ bfrag,
          unsigned short* __restrict__ mn2f) {
  __shared__ float sAW[COPM * CM], sBW[COPM * CM];
  __shared__ float sONW[CM], sONB[CM], sPNW[CM], sPNB[CM];
  int tid = threadIdx.x;
  for (int t = tid; t < COPM * CM; t += 256) { sAW[t] = oaw[t]; sBW[t] = obw[t]; }
  if (tid < CM) { sONW[tid] = onw[tid]; sONB[tid] = onb[tid];
                  sPNW[tid] = pnw[tid]; sPNB[tid] = pnb[tid]; }
  __syncthreads();

  int row = blockIdx.x * 256 + tid;            // row = s*256 + n
  const float* mr = m + (size_t)row * CM;
  float x[CM];
  float mu = 0.f;
  #pragma unroll
  for (int i = 0; i < CM; ++i) { x[i] = mr[i]; mu += x[i]; }
  mu *= (1.f / CM);
  float var = 0.f;
  #pragma unroll
  for (int i = 0; i < CM; ++i) { float d = x[i] - mu; var += d * d; }
  var *= (1.f / CM);
  float rinv = rsqrtf(var + 1e-5f);
  #pragma unroll
  for (int i = 0; i < CM; ++i) x[i] = (x[i] - mu) * rinv;  // normalized core

  float mask = msa_mask[row];
  int s = row >> 8, n = row & 255;

  for (int c = 0; c < COPM; ++c) {             // a/b projections (opm layernorm)
    float da = 0.f, db = 0.f;
    #pragma unroll
    for (int i = 0; i < CM; ++i) {
      float ln = x[i] * sONW[i] + sONB[i];
      da += sAW[c * CM + i] * ln;
      db += sBW[c * CM + i] * ln;
    }
    int mcol = n * COPM + c;                   // M/N index in 8192
    afrag[fragIdx(512, s, mcol)] = f2bf(da * mask);
    bfrag[fragIdx(512, s, mcol)] = f2bf(db * mask);
  }
  #pragma unroll
  for (int i = 0; i < CM; ++i)                 // mn2 A-fragments (pwa layernorm)
    mn2f[fragIdx(NROWS / 16, i, row)] = f2bf(x[i] * sPNW[i] + sPNB[i]);
}

// ====================== k_pwa_proj: WMMA rows x [m_w|g_w] -> v, g ======================
__global__ void __launch_bounds__(256)
k_pwa_proj(const unsigned short* __restrict__ mn2f,
           const unsigned short* __restrict__ mgw,
           unsigned short* __restrict__ vfrag, unsigned short* __restrict__ gfrag) {
  int wave = (blockIdx.x * 256 + threadIdx.x) >> 5;
  int lane = threadIdx.x & 31;
  int mT = wave >> 5;                          // 0..8191  (rows/16)
  int nT = wave & 31;                          // 0..31    (512 outputs /16)
  v8f acc = {0.f, 0.f, 0.f, 0.f, 0.f, 0.f, 0.f, 0.f};
  #pragma unroll
  for (int kT = 0; kT < 2; ++kT) {             // K = 64
    v16bf a = *(const v16bf*)(mn2f + (((size_t)kT * (NROWS / 16) + mT) << 9) + (lane << 4));
    v16bf b = *(const v16bf*)(mgw  + (((size_t)kT * 32 + nT) << 9) + (lane << 4));
    acc = wmma_bf16(a, b, acc);
  }
  int rowBase = mT * 16 + ((lane >> 4) << 3);
  int col = nT * 16 + (lane & 15);
  #pragma unroll
  for (int r = 0; r < 8; ++r) {
    int row = rowBase + r;                     // (s,j) row of m
    float val = acc[r];
    if (col < 256) {                           // v[s,j,h,c] -> per-head B-fragment (K=j)
      int s2 = row >> 8, j = row & 255;
      int h = col >> 5, c = col & 31;
      vfrag[(size_t)h * (256 * 16384) + fragIdx(1024, j, s2 * CHD + c)] = f2bf(val);
    } else {                                   // g = sigmoid -> A-fragment (K=h*32+c)
      float g = __builtin_amdgcn_rcpf(1.f + __expf(-val));
      gfrag[fragIdx(NROWS / 16, col - 256, row)] = f2bf(g);
    }
  }
}

// ====================== k_opm: fused OPM GEMM + o-proj + LN + bias ======================
__global__ void __launch_bounds__(256)
k_opm(const unsigned short* __restrict__ afrag, const unsigned short* __restrict__ bfrag,
      const unsigned short* __restrict__ owf, const float* __restrict__ rnum,
      const float* __restrict__ opm_o_b, const float* __restrict__ z_in,
      const float* __restrict__ pair_mask, const float* __restrict__ nzw,
      const float* __restrict__ nzb, const float* __restrict__ pwa_z_w,
      float* __restrict__ bias) {
  // 32 KiB buffer, used twice: (1) LDS-staged A tiles, (2) zt block as A-fragments
  __shared__ __align__(32) unsigned short sBuf[32 * 512];
  __shared__ float sZN[16 * CZ];                          // z_new block, 8 KiB
  __shared__ float sMu[16], sRinv[16];

  int tid = threadIdx.x, lane = tid & 31, w = tid >> 5;   // 8 waves
  int i = blockIdx.x >> 4;                                // 0..255
  int jBlk = blockIdx.x & 15;                             // 16 j's per block
  int j0 = jBlk * 16 + 2 * w, j1 = j0 + 1;                // this wave's pairs

  // ---- stage all 32 A-tiles (16 kTiles x 2 mTiles, 1 KiB each) into LDS ----
#if HAVE_ASYNC_LDS
  {
    int laneB = lane * 32;                                // per-lane 32B of each tile
    #pragma unroll
    for (int t = 0; t < 4; ++t) {                         // 4 tiles per wave
      int kT = w * 2 + (t >> 1);
      int mL = t & 1;
      const char* g =
          (const char*)(afrag + (((size_t)kT * 512 + (i * 2 + mL)) << 9)) + laneB;
      char* l = (char*)sBuf + ((size_t)(kT * 2 + mL) << 10) + laneB;
      __builtin_amdgcn_global_load_async_to_lds_b128(
          (gas_v4i)(void*)g, (las_v4i)(void*)l, 0, 0);
      __builtin_amdgcn_global_load_async_to_lds_b128(
          (gas_v4i)(void*)(g + 16), (las_v4i)(void*)(l + 16), 0, 0);
    }
  }
#if __has_builtin(__builtin_amdgcn_s_wait_asynccnt)
  __builtin_amdgcn_s_wait_asynccnt(0);
#else
  asm volatile("s_wait_asynccnt 0x0" ::: "memory");
#endif
#else
  for (int idx = tid; idx < 2048; idx += 256) {           // 2048 x 16B = 32 KiB
    int kT = idx >> 7, r = idx & 127, mL = r >> 6, inner = r & 63;
    ((uint4*)sBuf)[idx] =
        ((const uint4*)(afrag + (((size_t)kT * 512 + (i * 2 + mL)) << 9)))[inner];
  }
#endif
  __syncthreads();

  const v8f vzero = {0.f, 0.f, 0.f, 0.f, 0.f, 0.f, 0.f, 0.f};
  v8f acc[2][2][2];                                       // [pair][mLocal][nLocal]
  #pragma unroll
  for (int p = 0; p < 2; ++p)
    #pragma unroll
    for (int a = 0; a < 2; ++a)
      #pragma unroll
      for (int b = 0; b < 2; ++b) acc[p][a][b] = vzero;

  // ---- stage 1: zt[(i,c),(j,d)] = sum_s a * b ; K = 512, A from LDS, B from L2
  #pragma unroll 1
  for (int kT = 0; kT < 16; ++kT) {
    const unsigned short* ab = sBuf + ((kT * 2) << 9) + (lane << 4);
    const unsigned short* bb = bfrag + (((size_t)kT * 512) << 9) + (lane << 4);
    if (kT + 1 < 16)                                      // -> global_prefetch_b8
      __builtin_prefetch((const void*)(bb + (1ull << 18) + ((size_t)(j0 * 2) << 9)), 0, 0);
    v16bf a0  = *(const v16bf*)(ab);
    v16bf a1  = *(const v16bf*)(ab + 512);
    v16bf b00 = *(const v16bf*)(bb + ((size_t)(j0 * 2 + 0) << 9));
    v16bf b01 = *(const v16bf*)(bb + ((size_t)(j0 * 2 + 1) << 9));
    v16bf b10 = *(const v16bf*)(bb + ((size_t)(j1 * 2 + 0) << 9));
    v16bf b11 = *(const v16bf*)(bb + ((size_t)(j1 * 2 + 1) << 9));
    acc[0][0][0] = wmma_bf16(a0, b00, acc[0][0][0]);
    acc[0][0][1] = wmma_bf16(a0, b01, acc[0][0][1]);
    acc[0][1][0] = wmma_bf16(a1, b00, acc[0][1][0]);
    acc[0][1][1] = wmma_bf16(a1, b01, acc[0][1][1]);
    acc[1][0][0] = wmma_bf16(a0, b10, acc[1][0][0]);
    acc[1][0][1] = wmma_bf16(a0, b11, acc[1][0][1]);
    acc[1][1][0] = wmma_bf16(a1, b10, acc[1][1][0]);
    acc[1][1][1] = wmma_bf16(a1, b11, acc[1][1][1]);
  }
  __syncthreads();                                        // done reading A tiles

  // ---- stage zt/num into LDS as A-fragments (M = pair 0..15, K = c*32+d)
  #pragma unroll
  for (int p = 0; p < 2; ++p) {
    int j = j0 + p;
    float rn = rnum[i * 256 + j];
    int prow = 2 * w + p;
    #pragma unroll
    for (int mL = 0; mL < 2; ++mL)
      #pragma unroll
      for (int nL = 0; nL < 2; ++nL)
        #pragma unroll
        for (int r = 0; r < 8; ++r) {
          int c = mL * 16 + r + ((lane >> 4) << 3);
          int d = nL * 16 + (lane & 15);
          int k = c * 32 + d;                             // torch-style c-major flatten
          int ki = k & 31;
          int l2 = prow + (((ki >> 3) & 1) << 4);
          int hw = (ki & 7) + ((ki >> 4) << 3);
          sBuf[((k >> 5) << 9) + (l2 << 4) + hw] = f2bf(acc[p][mL][nL][r] * rn);
        }
  }
  __syncthreads();

  // ---- stage 2: z_opm = zt @ opm_o_w^T ; M=16 pairs, N=128 (wave w -> z-tile w), K=1024
  v8f zacc = vzero;
  #pragma unroll 1
  for (int kT = 0; kT < 32; ++kT) {
    v16bf a = *(const v16bf*)(sBuf + (kT << 9) + (lane << 4));    // LDS A-frag
    v16bf b = *(const v16bf*)(owf + (((size_t)kT * 8 + w) << 9) + (lane << 4));
    zacc = wmma_bf16(a, b, zacc);
  }
  {
    int zc = w * 16 + (lane & 15);
    int pBase = ((lane >> 4) << 3);
    #pragma unroll
    for (int r = 0; r < 8; ++r) {
      int p = pBase + r;
      int j = jBlk * 16 + p;
      sZN[p * CZ + zc] = zacc[r] + opm_o_b[zc] + z_in[((size_t)i * 256 + j) * CZ + zc];
    }
  }
  __syncthreads();

  // ---- layernorm stats over c_z per pair
  if (tid < 16) {
    float mu = 0.f;
    for (int zc = 0; zc < CZ; ++zc) mu += sZN[tid * CZ + zc];
    mu *= (1.f / CZ);
    float var = 0.f;
    for (int zc = 0; zc < CZ; ++zc) { float d = sZN[tid * CZ + zc] - mu; var += d * d; }
    var *= (1.f / CZ);
    sMu[tid] = mu; sRinv[tid] = rsqrtf(var + 1e-5f);
  }
  __syncthreads();

  // ---- bias[h,i,j] = zn . pwa_z_w[h] + (1-pair_mask)*(-1e6)
  if (tid < 128) {
    int p = tid >> 3, h = tid & 7;
    int j = jBlk * 16 + p;
    float mu = sMu[p], ri = sRinv[p];
    float acc2 = 0.f;
    for (int zc = 0; zc < CZ; ++zc) {
      float zn = (sZN[p * CZ + zc] - mu) * ri * nzw[zc] + nzb[zc];
      acc2 += zn * pwa_z_w[h * CZ + zc];
    }
    acc2 += (1.0f - pair_mask[i * 256 + j]) * (-1.0e6f);
    bias[((size_t)h * 256 + i) * 256 + j] = acc2;
  }
}

// ====================== k_softmax -> W in A-fragment layout ======================
__global__ void __launch_bounds__(256)
k_softmax(const float* __restrict__ bias, unsigned short* __restrict__ wfrag) {
  __shared__ float red[256];
  int hi = blockIdx.x;                         // h*256 + i
  int tid = threadIdx.x;                       // j
  float v = bias[(size_t)hi * 256 + tid];
  red[tid] = v; __syncthreads();
  for (int st = 128; st > 0; st >>= 1) {
    if (tid < st) red[tid] = fmaxf(red[tid], red[tid + st]);
    __syncthreads();
  }
  float mx = red[0]; __syncthreads();
  float e = __expf(v - mx);
  red[tid] = e; __syncthreads();
  for (int st = 128; st > 0; st >>= 1) {
    if (tid < st) red[tid] += red[tid + st];
    __syncthreads();
  }
  float wv = e * __builtin_amdgcn_rcpf(red[0]);
  int h = hi >> 8, i = hi & 255;
  wfrag[(size_t)h * 65536 + fragIdx(16, tid, i)] = f2bf(wv);   // M=i, K=j
}

// ====================== k_attn: per-head O = W x V ======================
__global__ void __launch_bounds__(256)
k_attn(const unsigned short* __restrict__ wfrag, const unsigned short* __restrict__ vfrag,
       unsigned short* __restrict__ ofrag) {
  int wave = (blockIdx.x * 256 + threadIdx.x) >> 5;
  int lane = threadIdx.x & 31;
  int h = wave >> 14;                          // 16*1024 wave-tiles per head
  int rem = wave & 16383;
  int mT = rem >> 10;                          // i tile (16)
  int nT = rem & 1023;                         // (s,c) tile (1024)
  const unsigned short* wf = wfrag + (size_t)h * 65536;
  const unsigned short* vf = vfrag + (size_t)h * (256 * 16384);
  v8f acc = {0.f, 0.f, 0.f, 0.f, 0.f, 0.f, 0.f, 0.f};
  #pragma unroll
  for (int kT = 0; kT < 8; ++kT) {             // K = j = 256
    v16bf a = *(const v16bf*)(wf + (((size_t)kT * 16 + mT) << 9) + (lane << 4));
    v16bf b = *(const v16bf*)(vf + (((size_t)kT * 1024 + nT) << 9) + (lane << 4));
    acc = wmma_bf16(a, b, acc);
  }
  int iBase = mT * 16 + ((lane >> 4) << 3);
  int np = nT * 16 + (lane & 15);
  int s2 = np >> 5, c = np & 31;
  #pragma unroll
  for (int r = 0; r < 8; ++r) {                // scatter to A-frag for final GEMM
    int row = s2 * 256 + (iBase + r);
    ofrag[fragIdx(NROWS / 16, h * CHD + c, row)] = f2bf(acc[r]);
  }
}

// ====================== k_final: out = (g*o) @ pwa_o_w^T ======================
__global__ void __launch_bounds__(256)
k_final(const unsigned short* __restrict__ ofrag, const unsigned short* __restrict__ gfrag,
        const unsigned short* __restrict__ powf, float* __restrict__ out) {
  int wave = (blockIdx.x * 256 + threadIdx.x) >> 5;
  int lane = threadIdx.x & 31;
  int mT = wave >> 2;                          // row tile (8192)
  int nT = wave & 3;                           // output tile (64/16)
  v8f acc = {0.f, 0.f, 0.f, 0.f, 0.f, 0.f, 0.f, 0.f};
  #pragma unroll
  for (int kT = 0; kT < 8; ++kT) {             // K = h*c = 256
    size_t off = (((size_t)kT * (NROWS / 16) + mT) << 9) + (lane << 4);
    v16bf o = *(const v16bf*)(ofrag + off);
    v16bf g = *(const v16bf*)(gfrag + off);
    v16bf a;
    #pragma unroll
    for (int e = 0; e < 16; ++e) a[e] = (__bf16)((float)o[e] * (float)g[e]);
    v16bf b = *(const v16bf*)(powf + (((size_t)kT * 4 + nT) << 9) + (lane << 4));
    acc = wmma_bf16(a, b, acc);
  }
  int rowBase = mT * 16 + ((lane >> 4) << 3);
  int col = nT * 16 + (lane & 15);
  #pragma unroll
  for (int r = 0; r < 8; ++r)
    out[(size_t)(rowBase + r) * CM + col] = acc[r];
}

// ====================== host launcher ======================
extern "C" void kernel_launch(void* const* d_in, const int* in_sizes, int n_in,
                              void* d_out, int out_size, void* d_ws, size_t ws_size,
                              hipStream_t stream) {
  const float* m        = (const float*)d_in[0];
  const float* z        = (const float*)d_in[1];
  const float* msa_mask = (const float*)d_in[2];
  const float* pair_mask= (const float*)d_in[3];
  const float* opm_nw   = (const float*)d_in[4];
  const float* opm_nb   = (const float*)d_in[5];
  const float* opm_aw   = (const float*)d_in[6];
  const float* opm_bw   = (const float*)d_in[7];
  const float* opm_ow   = (const float*)d_in[8];
  const float* opm_ob   = (const float*)d_in[9];
  const float* pwa_nmw  = (const float*)d_in[10];
  const float* pwa_nmb  = (const float*)d_in[11];
  const float* pwa_nzw  = (const float*)d_in[12];
  const float* pwa_nzb  = (const float*)d_in[13];
  const float* pwa_mw   = (const float*)d_in[14];
  const float* pwa_gw   = (const float*)d_in[15];
  const float* pwa_zw   = (const float*)d_in[16];
  const float* pwa_ow   = (const float*)d_in[17];

  if (ws_size < WS_TOTAL) return;              // needs ~228 MiB scratch

  char* ws = (char*)d_ws;
  unsigned short* AFRAG = (unsigned short*)(ws + OFF_AFRAG);
  unsigned short* BFRAG = (unsigned short*)(ws + OFF_BFRAG);
  unsigned short* MN2F  = (unsigned short*)(ws + OFF_MN2F);
  unsigned short* OWF   = (unsigned short*)(ws + OFF_OWF);
  unsigned short* MGW   = (unsigned short*)(ws + OFF_MGW);
  unsigned short* POWF  = (unsigned short*)(ws + OFF_POW);
  float*          RNUM  = (float*)(ws + OFF_RNUM);
  float*          BIAS  = (float*)(ws + OFF_BIAS);
  unsigned short* WFRG  = (unsigned short*)(ws + OFF_WFRG);
  unsigned short* VFRG  = (unsigned short*)(ws + OFF_VFRG);
  unsigned short* GFRG  = (unsigned short*)(ws + OFF_GFRG);
  unsigned short* OFRG  = (unsigned short*)(ws + OFF_OFRG);
  float* out = (float*)d_out;

  k_prep    <<<704,   256, 0, stream>>>(opm_ow, pwa_mw, pwa_gw, pwa_ow, OWF, MGW, POWF);
  k_num     <<<256,   256, 0, stream>>>(msa_mask, RNUM);
  k_stage_m <<<512,   256, 0, stream>>>(m, msa_mask, opm_nw, opm_nb, opm_aw, opm_bw,
                                        pwa_nmw, pwa_nmb, AFRAG, BFRAG, MN2F);
  k_pwa_proj<<<32768, 256, 0, stream>>>(MN2F, MGW, VFRG, GFRG);
  k_opm     <<<4096,  256, 0, stream>>>(AFRAG, BFRAG, OWF, RNUM, opm_ob, z, pair_mask,
                                        pwa_nzw, pwa_nzb, pwa_zw, BIAS);
  k_softmax <<<2048,  256, 0, stream>>>(BIAS, WFRG);
  k_attn    <<<16384, 256, 0, stream>>>(WFRG, VFRG, OFRG);
  k_final   <<<4096,  256, 0, stream>>>(OFRG, GFRG, POWF, out);
}
